// RopeSelfAttention_56418690400951
// MI455X (gfx1250) — matrix operand order repaired and checked by reading, not compiled, not run on hardware
//
#include <hip/hip_runtime.h>
#include <stdint.h>

#define NHEAD 16
#define HDIM  64
#define SEQL  2048
#define BATCH 4
#define EMB   1024
#define NEG_INF_F (-1000000000.0f)

typedef __attribute__((ext_vector_type(16))) __bf16 v16bf;
typedef __attribute__((ext_vector_type(8)))  float  v8f;

union BfPack { uint4 q[2]; v16bf v; };

// Native bf16 conversions (gfx1250 has hardware bf16; clang lowers the
// fptrunc/fpext to single v_cvt-class ops instead of a bit-twiddle chain).
__device__ __forceinline__ unsigned short f2bf(float f) {
    union { __bf16 h; unsigned short u; } x;
    x.h = (__bf16)f;
    return x.u;
}
__device__ __forceinline__ float bf2f(unsigned short s) {
    union { unsigned short u; __bf16 h; } x;
    x.u = s;
    return (float)x.h;
}

__device__ __forceinline__ v8f wmma_bf16(v16bf a, v16bf b, v8f c) {
    return __builtin_amdgcn_wmma_f32_16x16x32_bf16(false, a, false, b, (short)0, c, false, false);
}

// --- CDNA5 async global->LDS DMA (ASYNCcnt-tracked) ----------------------
// LDS address: flat pointers in the LDS aperture map to LDS by truncation
// to addr[31:0] (ISA 10.2), so a truncated generic __shared__ pointer is a
// valid wave-relative LDS byte address for the VDST operand.
__device__ __forceinline__ void async_copy_b128(void* lds, const void* gptr) {
    unsigned ldsa = (unsigned)(uintptr_t)lds;
    unsigned long long ga = (unsigned long long)(uintptr_t)gptr;
    asm volatile("global_load_async_to_lds_b128 %0, %1, off"
                 :: "v"(ldsa), "v"(ga) : "memory");
}
__device__ __forceinline__ void wait_async_le2() {
    asm volatile("s_wait_asynccnt 0x2" ::: "memory");
}
__device__ __forceinline__ void wait_async_0() {
    asm volatile("s_wait_asynccnt 0x0" ::: "memory");
}

// A operand tile (16x32 bf16, row-major M x K, stride in halves).
// ISA layout: lane m=lane&15; halves e<8 -> k=e+8g, e>=8 -> k=16+(e-8)+8g.
__device__ __forceinline__ v16bf load_A_tile(const unsigned short* base, int stride, int lane) {
    int g = lane >> 4, m = lane & 15;
    BfPack p;
    p.q[0] = *(const uint4*)(base + m * stride + 8 * g);
    p.q[1] = *(const uint4*)(base + m * stride + 16 + 8 * g);
    return p.v;
}
// B operand tile stored transposed in LDS as N x K row-major (stride in halves).
// ISA layout: lane holds column n=lane&15; halves e<8 -> k=e+8g, e>=8 -> k=16+(e-8)+8g.
__device__ __forceinline__ v16bf load_B_tile(const unsigned short* base, int stride, int lane) {
    int g = lane >> 4, n = lane & 15;
    BfPack p;
    p.q[0] = *(const uint4*)(base + n * stride + 8 * g);
    p.q[1] = *(const uint4*)(base + n * stride + 16 + 8 * g);
    return p.v;
}

// ---------------------------------------------------------------------------
// Kernel 1: QKV = hidden @ qkv_w^T + bias ; scatter to Q,K [B,H,L,D] and
// Vt [B,H,D,L] (all bf16). C block 128x128 per WG, wave tile 32x64.
// f32 sources need conversion, so use register-prefetch software pipelining:
// next k-step's global loads are issued before the current WMMA burst.
// ---------------------------------------------------------------------------
__global__ __launch_bounds__(256) void qkv_gemm_kernel(
    const float* __restrict__ hidden, const float* __restrict__ qkv_w,
    const float* __restrict__ qkv_b, unsigned short* __restrict__ Qd,
    unsigned short* __restrict__ Kd, unsigned short* __restrict__ Vt)
{
    __shared__ __align__(16) unsigned short lsA[128 * 40];
    __shared__ __align__(16) unsigned short lsB[128 * 40];

    const int tid = threadIdx.x, lane = tid & 31, wid = tid >> 5;
    const int wm = wid & 3, wn = wid >> 2;          // 4 x 2 wave grid
    const int blockN = blockIdx.x * 128, blockM = blockIdx.y * 128;
    const int g = lane >> 4, nn = lane & 15;

    v8f acc[2][4];
    const v8f vzero = {0.f,0.f,0.f,0.f,0.f,0.f,0.f,0.f};
#pragma unroll
    for (int i = 0; i < 2; i++)
#pragma unroll
        for (int j = 0; j < 4; j++) acc[i][j] = vzero;

    const int ldr = tid >> 1;
    const int lseg = (tid & 1) * 16;
    const float* aptr = hidden + (size_t)(blockM + ldr) * EMB + lseg;
    const float* bptr = qkv_w  + (size_t)(blockN + ldr) * EMB + lseg;

    float4 fA[4], fB[4];
#pragma unroll
    for (int i = 0; i < 4; i++) { fA[i] = ((const float4*)aptr)[i]; fB[i] = ((const float4*)bptr)[i]; }

    for (int k0 = 0; k0 < EMB; k0 += 32) {
        // store staged registers (previous compute finished at bottom barrier)
        {
            union { uint4 q[2]; unsigned short s[16]; } pk;
#pragma unroll
            for (int i = 0; i < 4; i++) {
                pk.s[4*i+0] = f2bf(fA[i].x); pk.s[4*i+1] = f2bf(fA[i].y);
                pk.s[4*i+2] = f2bf(fA[i].z); pk.s[4*i+3] = f2bf(fA[i].w);
            }
            *(uint4*)(lsA + ldr * 40 + lseg)     = pk.q[0];
            *(uint4*)(lsA + ldr * 40 + lseg + 8) = pk.q[1];
#pragma unroll
            for (int i = 0; i < 4; i++) {
                pk.s[4*i+0] = f2bf(fB[i].x); pk.s[4*i+1] = f2bf(fB[i].y);
                pk.s[4*i+2] = f2bf(fB[i].z); pk.s[4*i+3] = f2bf(fB[i].w);
            }
            *(uint4*)(lsB + ldr * 40 + lseg)     = pk.q[0];
            *(uint4*)(lsB + ldr * 40 + lseg + 8) = pk.q[1];
        }
        __syncthreads();

        if (k0 + 32 < EMB) {    // prefetch next k-step into registers
            const float4* an = (const float4*)(aptr + k0 + 32);
            const float4* bn = (const float4*)(bptr + k0 + 32);
#pragma unroll
            for (int i = 0; i < 4; i++) { fA[i] = an[i]; fB[i] = bn[i]; }
        }

        v16bf a0 = load_A_tile(lsA + (wm * 32) * 40, 40, lane);
        v16bf a1 = load_A_tile(lsA + (wm * 32 + 16) * 40, 40, lane);
#pragma unroll
        for (int ns = 0; ns < 4; ns++) {
            v16bf b = load_B_tile(lsB + (wn * 64 + ns * 16) * 40, 40, lane);
            acc[0][ns] = wmma_bf16(a0, b, acc[0][ns]);
            acc[1][ns] = wmma_bf16(a1, b, acc[1][ns]);
        }
        __syncthreads();
    }

    // epilogue: +bias, scatter into Q / K / Vt
#pragma unroll
    for (int ms = 0; ms < 2; ms++) {
#pragma unroll
        for (int ns = 0; ns < 4; ns++) {
            int n = blockN + wn * 64 + ns * 16 + nn;
            float bias = qkv_b[n];
            int which = n >> 10;
            int e = n & 1023;
            int h = e >> 6, d = e & 63;
#pragma unroll
            for (int r = 0; r < 8; r++) {
                int m = blockM + wm * 32 + ms * 16 + r + 8 * g;
                int b = m >> 11, l = m & (SEQL - 1);
                unsigned short val = f2bf(acc[ms][ns][r] + bias);
                if (which == 0)
                    Qd[(((size_t)(b * NHEAD + h)) * SEQL + l) * HDIM + d] = val;
                else if (which == 1)
                    Kd[(((size_t)(b * NHEAD + h)) * SEQL + l) * HDIM + d] = val;
                else
                    Vt[(((size_t)(b * NHEAD + h)) * HDIM + d) * SEQL + l] = val;
            }
        }
    }
}

// ---------------------------------------------------------------------------
// Kernel 2: in-place RoPE on Q and K ([B,H,L,D] bf16). One thread per (bh,l,pair).
// ---------------------------------------------------------------------------
__global__ __launch_bounds__(256) void rope_kernel(unsigned short* __restrict__ Qd,
                                                   unsigned short* __restrict__ Kd)
{
    int idx = blockIdx.x * 256 + threadIdx.x;       // B*H*L*32 threads exactly
    int i  = idx & 31;                              // rotation pair index
    int l  = (idx >> 5) & (SEQL - 1);
    int bh = idx >> 16;
    if (bh >= BATCH * NHEAD) return;

    float inv_freq = __expf(-0.2878231366242557f * (float)i);  // 10000^(-i/32)
    float ang = (float)l * inv_freq;
    float sn, cs;
    sincosf(ang, &sn, &cs);

    size_t base = (((size_t)bh) * SEQL + l) * HDIM;
    float q1 = bf2f(Qd[base + i]), q2 = bf2f(Qd[base + i + 32]);
    Qd[base + i]      = f2bf(q1 * cs - q2 * sn);
    Qd[base + i + 32] = f2bf(q2 * cs + q1 * sn);
    float k1 = bf2f(Kd[base + i]), k2 = bf2f(Kd[base + i + 32]);
    Kd[base + i]      = f2bf(k1 * cs - k2 * sn);
    Kd[base + i + 32] = f2bf(k2 * cs + k1 * sn);
}

// ---------------------------------------------------------------------------
// Kernel 3: flash-style attention. Block = (bh, 128 queries), wave = 16 queries.
// K/V/mask tiles double-buffered and staged with async global->LDS DMA;
// next block's copies run concurrently with the current block's WMMAs.
// ---------------------------------------------------------------------------
__global__ __launch_bounds__(256) void attn_kernel(
    const unsigned short* __restrict__ Qd, const unsigned short* __restrict__ Kd,
    const unsigned short* __restrict__ Vt, const int* __restrict__ kpm,
    unsigned short* __restrict__ Od)
{
    __shared__ __align__(16) unsigned short lsK[2][32 * 72];   // [key][d]  (N x K for S)
    __shared__ __align__(16) unsigned short lsV[2][64 * 40];   // [d][key]  (N x K for PV)
    __shared__ __align__(16) unsigned short lsP[8 * 16 * 40];  // per-wave P tiles
    __shared__ __align__(16) int lsM[2][32];

    const int bh = blockIdx.x, b = bh >> 4;
    const int tid = threadIdx.x, lane = tid & 31, wid = tid >> 5;
    const int g = lane >> 4, nn = lane & 15;
    const int q0 = blockIdx.y * 128 + wid * 16;

    const unsigned short* kbase = Kd + (size_t)bh * SEQL * HDIM;
    const unsigned short* vbase = Vt + (size_t)bh * HDIM * SEQL;
    unsigned short* lsPw = lsP + wid * 16 * 40;

    const int krow = tid >> 3, kseg = (tid & 7) * 8;
    const int vrow = tid >> 2, vseg = (tid & 3) * 8;

    auto stage = [&](int buf, int key0) {
        async_copy_b128(lsK[buf] + krow * 72 + kseg,
                        kbase + (size_t)(key0 + krow) * HDIM + kseg);
        async_copy_b128(lsV[buf] + vrow * 40 + vseg,
                        vbase + (size_t)vrow * SEQL + key0 + vseg);
        if (tid < 8)
            async_copy_b128(&lsM[buf][tid * 4], kpm + b * SEQL + key0 + tid * 4);
    };

    // Q A-tiles for d=0..31 and d=32..63 (global bf16, A layout)
    v16bf aq0, aq1;
    {
        const unsigned short* row = Qd + ((size_t)bh * SEQL + q0 + nn) * HDIM;
        BfPack p;
        p.q[0] = *(const uint4*)(row + 8 * g);
        p.q[1] = *(const uint4*)(row + 16 + 8 * g);
        aq0 = p.v;
        p.q[0] = *(const uint4*)(row + 32 + 8 * g);
        p.q[1] = *(const uint4*)(row + 48 + 8 * g);
        aq1 = p.v;
    }

    float qmask[8];
#pragma unroll
    for (int r = 0; r < 8; r++)
        qmask[r] = (kpm[b * SEQL + q0 + r + 8 * g] == 0) ? 1.f : 0.f;

    float mi[8], li[8];
#pragma unroll
    for (int r = 0; r < 8; r++) { mi[r] = -3.0e38f; li[r] = 0.f; }
    const v8f vzero = {0.f,0.f,0.f,0.f,0.f,0.f,0.f,0.f};
    v8f o[4];
#pragma unroll
    for (int j = 0; j < 4; j++) o[j] = vzero;

    stage(0, 0);                               // prologue DMA for block 0

    for (int key0 = 0, it = 0; key0 < SEQL; key0 += 32, ++it) {
        const int buf = it & 1;
        const bool havenext = (key0 + 32) < SEQL;
        if (havenext) stage(buf ^ 1, key0 + 32);   // overlap next block's DMA
        // Async loads complete in issue order: waiting until only the
        // next-block copies remain outstanding guarantees buf is resident.
        if (havenext) wait_async_le2(); else wait_async_0();
        __syncthreads();

        const unsigned short* lk = lsK[buf];
        const unsigned short* lv = lsV[buf];

        // S = Q K^T for two 16-key subtiles
        float sc[2][8];
#pragma unroll
        for (int t = 0; t < 2; t++) {
            v8f s = vzero;
            v16bf bk0 = load_B_tile(lk + (t * 16) * 72, 72, lane);       // d 0..31
            s = wmma_bf16(aq0, bk0, s);
            v16bf bk1 = load_B_tile(lk + (t * 16) * 72 + 32, 72, lane);  // d 32..63
            s = wmma_bf16(aq1, bk1, s);
            float kallow = (lsM[buf][t * 16 + nn] == 0) ? 1.f : 0.f;
#pragma unroll
            for (int r = 0; r < 8; r++)
                sc[t][r] = (kallow * qmask[r] != 0.f) ? s[r] * 0.125f : NEG_INF_F;
        }

        // online softmax per row (rows r+8g live in this 16-lane half-group)
#pragma unroll
        for (int r = 0; r < 8; r++) {
            float bmax = fmaxf(sc[0][r], sc[1][r]);
#pragma unroll
            for (int m = 1; m < 16; m <<= 1) bmax = fmaxf(bmax, __shfl_xor(bmax, m, 32));
            float mnew = fmaxf(mi[r], bmax);
            float corr = __expf(mi[r] - mnew);
            float p0 = __expf(sc[0][r] - mnew);
            float p1 = __expf(sc[1][r] - mnew);
            float psum = p0 + p1;
#pragma unroll
            for (int m = 1; m < 16; m <<= 1) psum += __shfl_xor(psum, m, 32);
            li[r] = li[r] * corr + psum;
            mi[r] = mnew;
#pragma unroll
            for (int j = 0; j < 4; j++) o[j][r] *= corr;
            lsPw[(r + 8 * g) * 40 + nn]      = f2bf(p0);
            lsPw[(r + 8 * g) * 40 + 16 + nn] = f2bf(p1);
        }

        // O += P V  (A = P from per-wave LDS, contraction over the 32 keys)
        v16bf ap = load_A_tile(lsPw, 40, lane);
#pragma unroll
        for (int j = 0; j < 4; j++) {
            v16bf bv = load_B_tile(lv + (j * 16) * 40, 40, lane);
            o[j] = wmma_bf16(ap, bv, o[j]);
        }
        __syncthreads();
    }

    // finalize: O /= l, store bf16 [B,H,L,D]
    unsigned short* ob = Od + ((size_t)bh * SEQL + q0) * HDIM;
#pragma unroll
    for (int r = 0; r < 8; r++) {
        float inv = 1.f / li[r];
#pragma unroll
        for (int j = 0; j < 4; j++)
            ob[(size_t)(r + 8 * g) * HDIM + j * 16 + nn] = f2bf(o[j][r] * inv);
    }
}

// ---------------------------------------------------------------------------
// Kernel 4: out = attn_out @ proj_w^T + proj_b  (gathers [B,H,L,D] -> [B,L,E]).
// A-operand is raw bf16 -> double-buffered async global->LDS DMA;
// B-operand (f32 proj_w) uses register prefetch + convert.
// ---------------------------------------------------------------------------
__global__ __launch_bounds__(256) void proj_gemm_kernel(
    const unsigned short* __restrict__ Od, const float* __restrict__ proj_w,
    const float* __restrict__ proj_b, float* __restrict__ out)
{
    __shared__ __align__(16) unsigned short lsA[2][128 * 40];
    __shared__ __align__(16) unsigned short lsB[128 * 40];

    const int tid = threadIdx.x, lane = tid & 31, wid = tid >> 5;
    const int wm = wid & 3, wn = wid >> 2;
    const int blockN = blockIdx.x * 128, blockM = blockIdx.y * 128;
    const int g = lane >> 4, nn = lane & 15;

    v8f acc[2][4];
    const v8f vzero = {0.f,0.f,0.f,0.f,0.f,0.f,0.f,0.f};
#pragma unroll
    for (int i = 0; i < 2; i++)
#pragma unroll
        for (int j = 0; j < 4; j++) acc[i][j] = vzero;

    const int ldr = tid >> 1;
    const int lseg = (tid & 1) * 16;

    auto stageA = [&](int buf, int k0) {   // 32-wide k block is inside one head
        int m = blockM + ldr;
        int bb = m >> 11, l = m & (SEQL - 1);
        int h = k0 >> 6, dofs = (k0 & 63) + lseg;
        const unsigned short* src =
            Od + (((size_t)(bb * NHEAD + h)) * SEQL + l) * HDIM + dofs;
        async_copy_b128(lsA[buf] + ldr * 40 + lseg,     src);
        async_copy_b128(lsA[buf] + ldr * 40 + lseg + 8, src + 8);
    };

    const float* bptr = proj_w + (size_t)(blockN + ldr) * EMB + lseg;
    float4 fB[4];
#pragma unroll
    for (int i = 0; i < 4; i++) fB[i] = ((const float4*)bptr)[i];

    stageA(0, 0);

    for (int k0 = 0, it = 0; k0 < EMB; k0 += 32, ++it) {
        const int buf = it & 1;
        {   // store staged B registers (previous compute done at bottom barrier)
            union { uint4 q[2]; unsigned short s[16]; } pk;
#pragma unroll
            for (int i = 0; i < 4; i++) {
                pk.s[4*i+0] = f2bf(fB[i].x); pk.s[4*i+1] = f2bf(fB[i].y);
                pk.s[4*i+2] = f2bf(fB[i].z); pk.s[4*i+3] = f2bf(fB[i].w);
            }
            *(uint4*)(lsB + ldr * 40 + lseg)     = pk.q[0];
            *(uint4*)(lsB + ldr * 40 + lseg + 8) = pk.q[1];
        }
        const bool havenext = (k0 + 32) < EMB;
        if (havenext) {
            stageA(buf ^ 1, k0 + 32);
            const float4* bn = (const float4*)(bptr + k0 + 32);
#pragma unroll
            for (int i = 0; i < 4; i++) fB[i] = bn[i];
        }
        if (havenext) wait_async_le2(); else wait_async_0();
        __syncthreads();

        v16bf a0 = load_A_tile(lsA[buf] + (wm * 32) * 40, 40, lane);
        v16bf a1 = load_A_tile(lsA[buf] + (wm * 32 + 16) * 40, 40, lane);
#pragma unroll
        for (int ns = 0; ns < 4; ns++) {
            v16bf b = load_B_tile(lsB + (wn * 64 + ns * 16) * 40, 40, lane);
            acc[0][ns] = wmma_bf16(a0, b, acc[0][ns]);
            acc[1][ns] = wmma_bf16(a1, b, acc[1][ns]);
        }
        __syncthreads();
    }

#pragma unroll
    for (int ms = 0; ms < 2; ms++) {
#pragma unroll
        for (int ns = 0; ns < 4; ns++) {
            int n = blockN + wn * 64 + ns * 16 + nn;
            float bias = proj_b[n];
#pragma unroll
            for (int r = 0; r < 8; r++) {
                int m = blockM + wm * 32 + ms * 16 + r + 8 * g;
                out[(size_t)m * EMB + n] = acc[ms][ns][r] + bias;
            }
        }
    }
}

// ---------------------------------------------------------------------------
extern "C" void kernel_launch(void* const* d_in, const int* in_sizes, int n_in,
                              void* d_out, int out_size, void* d_ws, size_t ws_size,
                              hipStream_t stream)
{
    (void)in_sizes; (void)n_in; (void)out_size; (void)ws_size;
    const float* hidden = (const float*)d_in[0];
    const int*   kpm    = (const int*)d_in[1];
    const float* qkv_w  = (const float*)d_in[2];
    const float* qkv_b  = (const float*)d_in[3];
    const float* proj_w = (const float*)d_in[4];
    const float* proj_b = (const float*)d_in[5];
    float* out = (float*)d_out;

    const size_t TEN = (size_t)BATCH * NHEAD * SEQL * HDIM;  // 8388608 elems
    unsigned short* ws = (unsigned short*)d_ws;
    unsigned short* Qd = ws;
    unsigned short* Kd = ws + TEN;
    unsigned short* Vt = ws + 2 * TEN;
    unsigned short* Od = ws + 3 * TEN;

    qkv_gemm_kernel<<<dim3(3072 / 128, (BATCH * SEQL) / 128), 256, 0, stream>>>(
        hidden, qkv_w, qkv_b, Qd, Kd, Vt);
    rope_kernel<<<dim3((BATCH * NHEAD * SEQL * 32) / 256), 256, 0, stream>>>(Qd, Kd);
    attn_kernel<<<dim3(BATCH * NHEAD, SEQL / 128), 256, 0, stream>>>(Qd, Kd, Vt, kpm, Od);
    proj_gemm_kernel<<<dim3(EMB / 128, (BATCH * SEQL) / 128), 256, 0, stream>>>(
        Od, proj_w, proj_b, out);
}
